// Encoder_21715354649023
// MI455X (gfx1250) — matrix operand order repaired and checked by reading, not compiled, not run on hardware
//
#include <hip/hip_runtime.h>
#include <hip/hip_bf16.h>

// Problem constants (match reference)
#define VOCAB 32000
#define EMB   64
#define HID   128
#define BATCH 256
#define TLEN  512

typedef __attribute__((ext_vector_type(8)))  float  v8f;
typedef __attribute__((ext_vector_type(16))) __bf16 v16bf;

// 8-dword view of a 16-element bf16 WMMA operand
union BV {
    v16bf    v;
    unsigned u[8];
    uint4    q[2];
};

// Native f32 -> bf16 conversion (v_cvt_*_bf16_f32 on gfx1250)
__device__ __forceinline__ unsigned short f2bf(float f) {
    __bf16 b = (__bf16)f;
    return __builtin_bit_cast(unsigned short, b);
}
__device__ __forceinline__ unsigned pack2(float a, float b) {
    return (unsigned)f2bf(a) | ((unsigned)f2bf(b) << 16);
}

__device__ __forceinline__ float fast_rcp(float x) {
    return __builtin_amdgcn_rcpf(x);
}
__device__ __forceinline__ float sigmoidf_fast(float x) {
    return fast_rcp(1.f + __expf(-x));
}
__device__ __forceinline__ float tanhf_fast(float x) {
#if __has_builtin(__builtin_amdgcn_tanhf)
    return __builtin_amdgcn_tanhf(x);   // V_TANH_F32 (gfx1250 trans op)
#else
    float e2 = __expf(-2.f * x);
    return (1.f - e2) * fast_rcp(1.f + e2);
#endif
}

// Fused: embedding gather + [h|x] @ [U;W] recurrent scan.
// Grid: BATCH/16 = 16 workgroups. Block: 256 threads = 8 wave32.
// Wave w owns output columns [16w, 16w+16). h and x tiles shared via LDS
// (ping-pong double buffer -> single barrier per timestep).
__global__ __launch_bounds__(256)
void gru_scan_fused(const int*   __restrict__ tokens,
                    const float* __restrict__ emb,
                    const float* __restrict__ W,      // [EMB][HID]
                    const float* __restrict__ U,      // [HID][HID]
                    const float* __restrict__ b_g,
                    const float* __restrict__ b_h,
                    const float* __restrict__ zeta,
                    const float* __restrict__ nu,
                    float*       __restrict__ out)    // [B*T*HID] ++ [B*HID]
{
    __shared__ unsigned short hbf[2][16][HID];  // bf16 hidden state tile
    __shared__ unsigned short xbf[2][16][EMB];  // bf16 gathered embedding tile

    const int tid  = threadIdx.x;
    const int wave = tid >> 5;
    const int lane = tid & 31;
    const int hi   = lane >> 4;        // which 16-lane half
    const int ln   = lane & 15;
    const int hi8  = hi * 8;
    const int N    = wave * 16 + ln;   // this lane's output column
    const int bbase = blockIdx.x * 16; // batch-row base for this workgroup

    // scalar gates (computed redundantly per thread, trivial cost)
    const float mix = sigmoidf_fast(zeta[0]);
    const float nus = sigmoidf_fast(nu[0]);
    const float bg  = b_g[N];
    const float bh  = b_h[N];

    // ---- Preload B operands: U rows 0..127 -> kb 0..3, W rows 0..63 -> kb 4..5.
    // bf16 B (32x16), contiguous k as in A: VGPR j, lane half hi: rows {16*hi + 2j, 16*hi + 2j + 1}, col = ln.
    BV Bk[6];
#pragma unroll
    for (int kb = 0; kb < 6; ++kb) {
        const float* src  = (kb < 4) ? U : W;
        const int    rb   = (kb < 4) ? 32 * kb : 32 * (kb - 4);
#pragma unroll
        for (int j = 0; j < 8; ++j) {
            const int k0 = rb + hi * 16 + 2 * j;
            Bk[kb].u[j] = pack2(src[k0 * HID + N], src[(k0 + 1) * HID + N]);
        }
    }

    // Gather mapping: each thread fills 4 consecutive EMB columns of one row.
    const int grow = tid >> 4;
    const int gcol = (tid & 15) * 4;

    // ---- init: h0 = 0, gather x for t=0 into buffer 0
#pragma unroll
    for (int i = 0; i < 4; ++i)
        ((unsigned*)&hbf[0][0][0])[tid + i * 256] = 0u;   // 1024 dwords total
    {
        const int tok = tokens[(bbase + grow) * TLEN + 0];
        const float4 e = *(const float4*)(emb + (size_t)tok * EMB + gcol);
        uint2 p; p.x = pack2(e.x, e.y); p.y = pack2(e.z, e.w);
        *(uint2*)&xbf[0][grow][gcol] = p;
    }
    __syncthreads();

    float hprev[8];
#pragma unroll
    for (int j = 0; j < 8; ++j) hprev[j] = 0.f;

    for (int t = 0; t < TLEN; ++t) {
        const int cur = t & 1, nxt = cur ^ 1;

        // Prefetch next step's embeddings into the other buffer
        // (global loads overlap the WMMA below).
        if (t + 1 < TLEN) {
            const int tok = tokens[(bbase + grow) * TLEN + (t + 1)];
            const float4 e = *(const float4*)(emb + (size_t)tok * EMB + gcol);
            uint2 p; p.x = pack2(e.x, e.y); p.y = pack2(e.z, e.w);
            *(uint2*)&xbf[nxt][grow][gcol] = p;
        }

        // pre = h @ U + x @ W   (fp32 accumulate, bf16 inputs)
        v8f acc;
#pragma unroll
        for (int j = 0; j < 8; ++j) acc[j] = 0.f;

#pragma unroll
        for (int kb = 0; kb < 4; ++kb) {          // h part, K = 0..127
            BV a;
            a.q[0] = *(const uint4*)&hbf[cur][ln][kb * 32 + hi * 16];
            a.q[1] = *(const uint4*)&hbf[cur][ln][kb * 32 + hi * 16 + 8];
            acc = __builtin_amdgcn_wmma_f32_16x16x32_bf16(
                false, a.v, false, Bk[kb].v, (short)0, acc, false, false);
        }
#pragma unroll
        for (int kb = 0; kb < 2; ++kb) {          // x part, K = 128..191
            BV a;
            a.q[0] = *(const uint4*)&xbf[cur][ln][kb * 32 + hi * 16];
            a.q[1] = *(const uint4*)&xbf[cur][ln][kb * 32 + hi * 16 + 8];
            acc = __builtin_amdgcn_wmma_f32_16x16x32_bf16(
                false, a.v, false, Bk[4 + kb].v, (short)0, acc, false, false);
        }

        // gates + state update; C layout: VGPR j -> M = j + 8*hi, col = N
#pragma unroll
        for (int j = 0; j < 8; ++j) {
            const int M  = j + hi8;
            const float pre = acc[j];
            const float z   = sigmoidf_fast(pre + bg);
            const float c   = tanhf_fast(pre + bh);
            const float hn  = z * hprev[j] + (mix * (1.f - z) + nus) * c;
            hprev[j] = hn;
            out[((size_t)(bbase + M) * TLEN + t) * HID + N] = hn;
            hbf[nxt][M][N] = f2bf(hn);
        }
        __syncthreads();   // publish hbf[nxt]/xbf[nxt] before next step reads them
    }

    // final state after the scan
    float* state = out + (size_t)BATCH * TLEN * HID;
#pragma unroll
    for (int j = 0; j < 8; ++j) {
        const int M = j + hi8;
        state[(size_t)(bbase + M) * HID + N] = hprev[j];
    }
}

extern "C" void kernel_launch(void* const* d_in, const int* in_sizes, int n_in,
                              void* d_out, int out_size, void* d_ws, size_t ws_size,
                              hipStream_t stream) {
    (void)in_sizes; (void)n_in; (void)d_ws; (void)ws_size; (void)out_size;
    const int*   tokens = (const int*)  d_in[0];
    const float* emb    = (const float*)d_in[1];
    const float* W      = (const float*)d_in[2];
    const float* U      = (const float*)d_in[3];
    const float* b_g    = (const float*)d_in[4];
    const float* b_h    = (const float*)d_in[5];
    const float* zeta   = (const float*)d_in[6];
    const float* nu     = (const float*)d_in[7];
    float* out = (float*)d_out;

    gru_scan_fused<<<BATCH / 16, 256, 0, stream>>>(
        tokens, emb, W, U, b_g, b_h, zeta, nu, out);
}
